// DyRAggregator_57612691309382
// MI455X (gfx1250) — matrix operand order repaired and checked by reading, not compiled, not run on hardware
//
#include <hip/hip_runtime.h>
#include <math.h>

typedef __attribute__((ext_vector_type(16))) __bf16 v16bf;
typedef __attribute__((ext_vector_type(8)))  float  v8f;

#define B_BATCH 16384
#define N_CAPS  32
#define DIN     128
#define DOUT    256
#define RITERS  3

__device__ __forceinline__ unsigned short f2bf_rne(float f) {
  unsigned u = __float_as_uint(f);
  u += 0x7FFFu + ((u >> 16) & 1u);           // round-to-nearest-even to bf16
  return (unsigned short)(u >> 16);
}
__device__ __forceinline__ float bf2f(unsigned short h) {
  return __uint_as_float(((unsigned)h) << 16);
}

// --- Prep: swizzle W (128x256 f32, row-major) into WMMA B-fragment layout,
// split into bf16 hi / lo halves. Fragment fi = ct*4 + kc covers cols
// [ct*16, ct*16+16) x K [kc*32, kc*32+32). Lane l holds column N = l%16;
// elements e: K = (e&7) + 16*(e>=8) + 8*(lane>=16).
__global__ void dyr_wswizzle(const float* __restrict__ W, unsigned short* __restrict__ wsw) {
  int idx = blockIdx.x * blockDim.x + threadIdx.x;   // 0 .. 16*4*32*16-1
  if (idx >= 16 * 4 * 32 * 16) return;
  int e  = idx & 15;
  int l  = (idx >> 4) & 31;
  int kc = (idx >> 9) & 3;
  int ct = idx >> 11;
  int n = ct * 16 + (l & 15);
  int k = kc * 32 + (e & 7) + ((e >> 3) << 4) + ((l >> 4) << 3);
  float val = W[k * DOUT + n];
  unsigned short hi = f2bf_rne(val);
  unsigned short lo = f2bf_rne(val - bf2f(hi));
  wsw[idx]         = hi;
  wsw[32768 + idx] = lo;
}

// --- Fused kernel: one workgroup (256 threads = 8 wave32) per batch element.
__global__ __launch_bounds__(256)
void dyr_fused(const float* __restrict__ embeds, const float* __restrict__ weights,
               const float* __restrict__ bias, const unsigned short* __restrict__ wsw,
               float* __restrict__ out) {
  __shared__ __align__(32) unsigned short aHi[2][4][32][16];  // A frags (rt,kc)
  __shared__ __align__(32) unsigned short aLo[2][4][32][16];
  __shared__ float uhat[32][264];   // padded stride to dodge bank conflicts
  __shared__ float pred[32][8];
  __shared__ float bij[32];
  __shared__ float cbuf[32];
  __shared__ float vbuf[256];
  __shared__ float wred[8];
  __shared__ float sscale;

  const int b    = blockIdx.x;
  const int tid  = threadIdx.x;
  const int lane = tid & 31;
  const int wave = tid >> 5;

  // Phase 1: stage embeds[b] as bf16 hi/lo A-fragments in LDS.
  // 16-bit A 16x32 layout: K = e + 8*(e>=8) + 8*(lane>=16).
  {
    int rt = wave >> 2, kc = wave & 3;                // wave -> fragment
    int n  = rt * 16 + (lane & 15);
    const float* src = embeds + ((size_t)b * N_CAPS + n) * DIN + kc * 32;
    int kadd = (lane >> 4) << 3;
    #pragma unroll
    for (int e = 0; e < 16; ++e) {
      int K = e + ((e >> 3) << 3) + kadd;
      float val = src[K];
      unsigned short hi = f2bf_rne(val);
      aHi[rt][kc][lane][e] = hi;
      aLo[rt][kc][lane][e] = f2bf_rne(val - bf2f(hi));
    }
  }
  __syncthreads();

  // Phase 2: GEMM via split-bf16 WMMA (hi*hi + hi*lo + lo*hi), fp32 accum.
  for (int ci = 0; ci < 2; ++ci) {
    int ct = wave * 2 + ci;
    v16bf bhi[4], blo[4];
    #pragma unroll
    for (int kc = 0; kc < 4; ++kc) {
      size_t off = (size_t)(ct * 4 + kc) * 512 + (size_t)lane * 16;
      bhi[kc] = *(const v16bf*)(wsw + off);
      blo[kc] = *(const v16bf*)(wsw + 32768 + off);
    }
    float bval = bias[ct * 16 + (lane & 15)];
    #pragma unroll
    for (int rt = 0; rt < 2; ++rt) {
      v8f acc = {};
      #pragma unroll
      for (int kc = 0; kc < 4; ++kc) {
        const v16bf ahi = *(const v16bf*)&aHi[rt][kc][lane][0];
        const v16bf alo = *(const v16bf*)&aLo[rt][kc][lane][0];
        acc = __builtin_amdgcn_wmma_f32_16x16x32_bf16(false, ahi, false, bhi[kc], (short)0, acc, false, false);
        acc = __builtin_amdgcn_wmma_f32_16x16x32_bf16(false, ahi, false, blo[kc], (short)0, acc, false, false);
        acc = __builtin_amdgcn_wmma_f32_16x16x32_bf16(false, alo, false, bhi[kc], (short)0, acc, false, false);
      }
      // C/D layout: VGPR j -> row j (lanes 0-15) / j+8 (lanes 16-31), col = lane%16
      int col   = ct * 16 + (lane & 15);
      int rbase = rt * 16 + ((lane >> 4) << 3);
      #pragma unroll
      for (int j = 0; j < 8; ++j) uhat[rbase + j][col] = acc[j] + bval;
    }
  }
  __syncthreads();

  // Phase 3: row-normalize uhat (32 rows x 256; 8 threads per row).
  {
    int row = tid >> 3, sub = tid & 7;
    float ss = 0.f;
    for (int c0 = sub * 32; c0 < sub * 32 + 32; ++c0) { float x = uhat[row][c0]; ss += x * x; }
    pred[row][sub] = ss;
  }
  __syncthreads();
  if (tid < 32) {
    float ss = 0.f;
    #pragma unroll
    for (int s2 = 0; s2 < 8; ++s2) ss += pred[tid][s2];
    cbuf[tid] = 1.0f / fmaxf(sqrtf(ss), 1e-12f);   // temp: inverse norm per row
  }
  __syncthreads();
  {
    int row = tid >> 3, sub = tid & 7;
    float inv = cbuf[row];
    for (int c0 = sub * 32; c0 < sub * 32 + 32; ++c0) uhat[row][c0] *= inv;
  }
  if (tid < 32) bij[tid] = weights[(size_t)b * N_CAPS + tid];
  __syncthreads();

  // Phase 4: dynamic routing, fully in LDS.
  float v_d = 0.f;
  for (int r = 0; r < RITERS; ++r) {
    if (tid == 0) {
      float m = bij[0];
      for (int n2 = 1; n2 < 32; ++n2) m = fmaxf(m, bij[n2]);
      float ssum = 0.f;
      for (int n2 = 0; n2 < 32; ++n2) { float e2 = __expf(bij[n2] - m); cbuf[n2] = e2; ssum += e2; }
      float inv = 32.0f / ssum;
      for (int n2 = 0; n2 < 32; ++n2) cbuf[n2] *= inv;
    }
    __syncthreads();
    float s = 0.f;                               // this thread's output dim d = tid
    #pragma unroll 8
    for (int n2 = 0; n2 < 32; ++n2) s += cbuf[n2] * uhat[n2][tid];
    float t = s * s;                             // block reduce sum(s^2)
    #pragma unroll
    for (int off = 16; off > 0; off >>= 1) t += __shfl_down(t, off, 32);
    if (lane == 0) wred[wave] = t;
    __syncthreads();
    if (tid == 0) {
      float sq = 0.f;
      #pragma unroll
      for (int w2 = 0; w2 < 8; ++w2) sq += wred[w2];
      sscale = sq / (1.0f + sq) * __frsqrt_rn(sq + 1e-9f);   // squash scale
    }
    __syncthreads();
    v_d = sscale * s;
    vbuf[tid] = v_d;
    __syncthreads();
    if (r < RITERS - 1) {                        // agreement: uhat[n,:]·v
      int row = tid >> 3, sub = tid & 7;
      float p = 0.f;
      for (int c0 = sub * 32; c0 < sub * 32 + 32; ++c0) p += uhat[row][c0] * vbuf[c0];
      pred[row][sub] = p;
      __syncthreads();
      if (tid < 32) {
        float a = 0.f;
        #pragma unroll
        for (int s2 = 0; s2 < 8; ++s2) a += pred[tid][s2];
        bij[tid] += a;
      }
      __syncthreads();
    }
  }

  out[(size_t)b * DOUT + tid] = v_d;                                   // poses
  if (tid < 32) out[(size_t)B_BATCH * DOUT + (size_t)b * N_CAPS + tid] = cbuf[tid]; // c_out
}

extern "C" void kernel_launch(void* const* d_in, const int* in_sizes, int n_in,
                              void* d_out, int out_size, void* d_ws, size_t ws_size,
                              hipStream_t stream) {
  (void)in_sizes; (void)n_in; (void)out_size; (void)ws_size;
  const float* embeds  = (const float*)d_in[0];
  const float* weights = (const float*)d_in[1];
  const float* W       = (const float*)d_in[2];
  const float* bias    = (const float*)d_in[3];
  float* out = (float*)d_out;
  unsigned short* wsw = (unsigned short*)d_ws;   // 128 KB used

  dyr_wswizzle<<<(16 * 4 * 32 * 16 + 255) / 256, 256, 0, stream>>>(W, wsw);
  dyr_fused<<<B_BATCH, 256, 0, stream>>>(embeds, weights, bias, wsw, out);
}
